// MambaSimple_57011395887156
// MI455X (gfx1250) — matrix operand order repaired and checked by reading, not compiled, not run on hardware
//
#include <hip/hip_runtime.h>
#include <hip/hip_bf16.h>

typedef __attribute__((ext_vector_type(16))) _Float16 v16h;
typedef __attribute__((ext_vector_type(8)))  _Float16 v8h;
typedef __attribute__((ext_vector_type(8)))  float    v8f;

#define B_   16
#define L_   2048
#define CIN  7
#define DM   128
#define DI   256      // d_inner
#define NST  32       // d_state (D_FF)
#define DTR  8        // dt_rank
#define XDBL (DTR + 2*NST)   // 72
#define NLAY 2

// ---------------------------------------------------------------- stats
__global__ void stats_kernel(const float* __restrict__ x, float* __restrict__ mean,
                             float* __restrict__ stdv) {
    int bc = blockIdx.x;                 // 0..B*CIN-1
    int b = bc / CIN, c = bc % CIN;
    const float* p = x + (size_t)b * L_ * CIN + c;
    float s = 0.f, ss = 0.f;
    for (int t = threadIdx.x; t < L_; t += blockDim.x) {
        float v = p[(size_t)t * CIN];
        s += v; ss += v * v;
    }
    __shared__ float rs[256], rss[256];
    rs[threadIdx.x] = s; rss[threadIdx.x] = ss;
    __syncthreads();
    for (int o = 128; o > 0; o >>= 1) {
        if (threadIdx.x < o) { rs[threadIdx.x] += rs[threadIdx.x+o]; rss[threadIdx.x] += rss[threadIdx.x+o]; }
        __syncthreads();
    }
    if (threadIdx.x == 0) {
        float m = rs[0] / (float)L_;
        float var = rss[0] / (float)L_ - m * m;
        mean[bc] = m;
        stdv[bc] = sqrtf(var + 1e-5f);
    }
}

// ----------------------------------------- embedding: token conv + time + PE
__global__ void embed_kernel(const float* __restrict__ x, const float* __restrict__ xmark,
                             const float* __restrict__ tokw, const float* __restrict__ timew,
                             const float* __restrict__ mean, const float* __restrict__ stdv,
                             float* __restrict__ h) {
    int idx = blockIdx.x * blockDim.x + threadIdx.x;   // b*L*DM elements
    int d = idx & (DM - 1);
    int t = (idx >> 7) & (L_ - 1);
    int b = idx >> 18;
    int tm = (t + L_ - 1) & (L_ - 1);
    int tp = (t + 1) & (L_ - 1);
    float acc = 0.f;
    for (int c = 0; c < CIN; ++c) {
        float m = mean[b*CIN + c];
        float rs = 1.f / stdv[b*CIN + c];
        float xa = (x[((size_t)(b*L_ + tm))*CIN + c] - m) * rs;
        float xb = (x[((size_t)(b*L_ + t ))*CIN + c] - m) * rs;
        float xc = (x[((size_t)(b*L_ + tp))*CIN + c] - m) * rs;
        const float* w = tokw + (d*CIN + c) * 3;
        acc += xa * w[0] + xb * w[1] + xc * w[2];
    }
    const float* xm = xmark + ((size_t)(b*L_ + t)) * 4;
    for (int m2 = 0; m2 < 4; ++m2) acc += xm[m2] * timew[m2*DM + d];
    int p = d >> 1;
    float freq = __expf(-9.2103403719761836f * (float)(2*p) / (float)DM);
    float ang  = (float)t * freq;
    acc += (d & 1) ? __cosf(ang) : __sinf(ang);
    h[idx] = acc;
}

// ---------------------------------------------------- RMSNorm -> f16 staging
__global__ void rmsnorm_f16_kernel(const float* __restrict__ h, const float* __restrict__ nw,
                                   _Float16* __restrict__ out) {
    int tok = blockIdx.x;
    int tid = threadIdx.x;          // 128
    float v = h[(size_t)tok * DM + tid];
    __shared__ float red[DM];
    red[tid] = v * v;
    __syncthreads();
    for (int o = 64; o > 0; o >>= 1) { if (tid < o) red[tid] += red[tid+o]; __syncthreads(); }
    float scale = rsqrtf(red[0] / (float)DM + 1e-5f);
    out[(size_t)tok * DM + tid] = (_Float16)(v * scale * nw[tid]);
}

// ------------- pack W[K,N] (f32, k-major) into WMMA B-fragment f16 layout
// layout: [(ntile*Ksteps + ks)*32 + lane][16 f16]; B lanes 0-15: K=0..7,16..23 of
// the 32-K step, lanes 16-31: K=8..15,24..31; n = ntile*16 + (lane&15)
__global__ void pack_w_kernel(const float* __restrict__ W, _Float16* __restrict__ out,
                              int K, int N, int total) {
    int idx = blockIdx.x * blockDim.x + threadIdx.x;
    if (idx >= total) return;
    int lane = idx & 31;
    int ksteps = K >> 5;
    int ks = (idx >> 5) % ksteps;
    int nt = (idx >> 5) / ksteps;
    int hf = lane >> 4, nn = lane & 15;
    int n = nt * 16 + nn;
    _Float16* o = out + (size_t)idx * 16;
    for (int j = 0; j < 16; ++j) {
        int k = ks * 32 + (j < 8 ? hf * 8 + j : 16 + hf * 8 + (j - 8));
        o[j] = (_Float16)W[(size_t)k * N + n];
    }
}

// ----------------------------------------------------------- WMMA GEMM
// C[M,N] (+)= A[M,K](f16,row-major) * Wfrag ; one wave per 16x16 tile
__global__ void gemm_wmma_kernel(const _Float16* __restrict__ A, const _Float16* __restrict__ Wf,
                                 float* __restrict__ C, int K, int N, int addres) {
    int lane = threadIdx.x & 31;
    int wave = blockIdx.x * (blockDim.x >> 5) + (threadIdx.x >> 5);
    int ntiles = N >> 4;
    int mtile = wave / ntiles;
    int ntile = wave % ntiles;
    int hf = lane >> 4, nn = lane & 15;
    int row = mtile * 16 + nn;               // A fragment row for this lane
    int ksteps = K >> 5;
    v8f c = {};
    const _Float16* arow = A + (size_t)row * K + hf * 8;
    const _Float16* wp   = Wf + (size_t)ntile * ksteps * 32 * 16 + (size_t)lane * 16;
    for (int ks = 0; ks < ksteps; ++ks) {
        union { v16h v; v8h h[2]; } a;
        a.h[0] = *(const v8h*)(arow);        // K pairs 0..7 of lane's half
        a.h[1] = *(const v8h*)(arow + 16);   // K pairs 16..23 of lane's half
        v16h bfrag = *(const v16h*)(wp);
        c = __builtin_amdgcn_wmma_f32_16x16x32_f16(false, a.v, false, bfrag,
                                                   (short)0, c, false, false);
        arow += 32;
        wp   += 32 * 16;
    }
    // D layout: vgpr v -> row mtile*16 + v + 8*hf, col ntile*16 + nn
    float* crow = C + (size_t)(mtile * 16 + hf * 8) * N + ntile * 16 + nn;
    for (int v = 0; v < 8; ++v) {
        float* p = crow + (size_t)v * N;
        if (addres) *p += c[v]; else *p = c[v];
    }
}

// ------------------------------------------- depthwise causal conv + SiLU
__global__ void conv_silu_kernel(const float* __restrict__ xr, const float* __restrict__ cw,
                                 const float* __restrict__ cb, float* __restrict__ xc) {
    int idx = blockIdx.x * blockDim.x + threadIdx.x;   // B*L*DI
    int d = idx & (DI - 1);
    int t = (idx >> 8) & (L_ - 1);
    int bt = idx >> 8;                                  // b*L + t
    float acc = cb[d];
    const float* w = cw + d * 4;
    for (int k = 0; k < 4; ++k) {
        int tt = t - 3 + k;
        if (tt >= 0) acc += xr[((size_t)(bt - 3 + k)) * (2*DI) + d] * w[k];
    }
    float sig = 1.f / (1.f + __expf(-acc));
    xc[idx] = acc * sig;
}

// ----------------------------- x_proj (K=256 -> 72) and delta = softplus(...)
__global__ void xproj_delta_kernel(const float* __restrict__ xc, const float* __restrict__ xpw,
                                   const float* __restrict__ dtw, const float* __restrict__ dtb,
                                   float* __restrict__ xdbl, float* __restrict__ delta) {
    int tok = blockIdx.x;
    int tid = threadIdx.x;   // 256
    __shared__ float xl[DI];
    __shared__ float dl[XDBL];
    xl[tid] = xc[(size_t)tok * DI + tid];
    __syncthreads();
    if (tid < XDBL) {
        float acc = 0.f;
        for (int k = 0; k < DI; ++k) acc += xl[k] * xpw[(size_t)k * XDBL + tid];
        dl[tid] = acc;
        xdbl[(size_t)tok * XDBL + tid] = acc;
    }
    __syncthreads();
    float s = dtb[tid];
    for (int r = 0; r < DTR; ++r) s += dl[r] * dtw[r * DI + tid];
    delta[(size_t)tok * DI + tid] = (s > 20.f) ? s : __logf(1.f + __expf(s));
}

// ---------------------- selective scan: one wave32 per (b,d), lane = state n
__global__ void scan_kernel(const float* __restrict__ delta, const float* __restrict__ u,
                            const float* __restrict__ xdbl, const float* __restrict__ xr,
                            const float* __restrict__ alog, const float* __restrict__ Dp,
                            _Float16* __restrict__ yg) {
    int lane = threadIdx.x & 31;
    int wave = blockIdx.x * (blockDim.x >> 5) + (threadIdx.x >> 5);
    int b = wave >> 8;           // / DI
    int d = wave & (DI - 1);
    float An = -__expf(alog[d * NST + lane]);
    float Dd = Dp[d];
    float x = 0.f;
    size_t tok = (size_t)b * L_;
    for (int t = 0; t < L_; ++t, ++tok) {
        float dv = delta[tok * DI + d];
        float uv = u[tok * DI + d];
        float Bv = xdbl[tok * XDBL + DTR + lane];
        float Cv = xdbl[tok * XDBL + DTR + NST + lane];
        float dA = __expf(dv * An);
        x = dA * x + (dv * uv) * Bv;
        float acc = x * Cv;
        acc += __shfl_xor(acc, 16, 32);
        acc += __shfl_xor(acc, 8, 32);
        acc += __shfl_xor(acc, 4, 32);
        acc += __shfl_xor(acc, 2, 32);
        acc += __shfl_xor(acc, 1, 32);
        if (lane == 0) {
            float res = xr[tok * (2*DI) + DI + d];
            float sil = res / (1.f + __expf(-res));
            yg[tok * DI + d] = (_Float16)((acc + uv * Dd) * sil);
        }
    }
}

// -------------------------- final RMSNorm + head (128->7) + de-normalize
__global__ void final_kernel(const float* __restrict__ h, const float* __restrict__ onw,
                             const float* __restrict__ ow, const float* __restrict__ mean,
                             const float* __restrict__ stdv, float* __restrict__ out) {
    int tok = blockIdx.x;
    int tid = threadIdx.x;   // 128
    int b = tok / L_;
    __shared__ float hn[DM];
    __shared__ float red[DM];
    float v = h[(size_t)tok * DM + tid];
    red[tid] = v * v;
    __syncthreads();
    for (int o = 64; o > 0; o >>= 1) { if (tid < o) red[tid] += red[tid+o]; __syncthreads(); }
    float scale = rsqrtf(red[0] / (float)DM + 1e-5f);
    hn[tid] = v * scale * onw[tid];
    __syncthreads();
    if (tid < CIN) {
        float acc = 0.f;
        for (int k = 0; k < DM; ++k) acc += hn[k] * ow[k * CIN + tid];
        out[(size_t)tok * CIN + tid] = acc * stdv[b*CIN + tid] + mean[b*CIN + tid];
    }
}

extern "C" void kernel_launch(void* const* d_in, const int* in_sizes, int n_in,
                              void* d_out, int out_size, void* d_ws, size_t ws_size,
                              hipStream_t stream) {
    (void)in_sizes; (void)n_in; (void)out_size; (void)ws_size;
    const float* x_enc      = (const float*)d_in[0];
    const float* x_mark     = (const float*)d_in[1];
    const float* tok_w      = (const float*)d_in[2];
    const float* time_w     = (const float*)d_in[3];
    const float* norm_w     = (const float*)d_in[4];
    const float* in_proj_w  = (const float*)d_in[5];
    const float* conv_w     = (const float*)d_in[6];
    const float* conv_b     = (const float*)d_in[7];
    const float* x_proj_w   = (const float*)d_in[8];
    const float* dt_proj_w  = (const float*)d_in[9];
    const float* dt_proj_b  = (const float*)d_in[10];
    const float* A_log      = (const float*)d_in[11];
    const float* D_param    = (const float*)d_in[12];
    const float* out_proj_w = (const float*)d_in[13];
    const float* out_norm_w = (const float*)d_in[14];
    const float* out_w      = (const float*)d_in[15];

    char* wsp = (char*)d_ws;
    auto alloc = [&](size_t bytes) -> void* {
        void* p = (void*)wsp;
        wsp += (bytes + 255) & ~(size_t)255;
        return p;
    };
    const size_t NT = (size_t)B_ * L_;                 // 32768 tokens
    float*    mean  = (float*)alloc((size_t)B_ * CIN * 4);
    float*    stdv  = (float*)alloc((size_t)B_ * CIN * 4);
    float*    h     = (float*)alloc(NT * DM * 4);
    float*    xr    = (float*)alloc(NT * 2 * DI * 4);  // [xi | res]
    float*    xc    = (float*)alloc(NT * DI * 4);      // conv+silu out (u)
    float*    xdbl  = (float*)alloc(NT * XDBL * 4);    // [dt | B | C]
    float*    delta = (float*)alloc(NT * DI * 4);
    _Float16* hn16  = (_Float16*)alloc(NT * DM * 2);
    _Float16* yg16  = (_Float16*)alloc(NT * DI * 2);
    _Float16* wfrag = (_Float16*)alloc((size_t)DM * 2 * DI * 2);  // max pack

    stats_kernel<<<B_ * CIN, 256, 0, stream>>>(x_enc, mean, stdv);
    embed_kernel<<<(B_ * L_ * DM) / 256, 256, 0, stream>>>(
        x_enc, x_mark, tok_w, time_w, mean, stdv, h);

    for (int l = 0; l < NLAY; ++l) {
        // in_proj: pack W[128,512] then GEMM [NT x 128]*[128 x 512] -> xr
        int totA = (512/16) * (DM/32) * 32;            // 4096
        pack_w_kernel<<<(totA + 255) / 256, 256, 0, stream>>>(
            in_proj_w + (size_t)l * DM * 2 * DI, wfrag, DM, 2 * DI, totA);
        rmsnorm_f16_kernel<<<(int)NT, DM, 0, stream>>>(h, norm_w + l * DM, hn16);
        gemm_wmma_kernel<<<(int)((NT/16) * (2*DI/16) / 8), 256, 0, stream>>>(
            hn16, wfrag, xr, DM, 2 * DI, 0);

        conv_silu_kernel<<<(int)(NT * DI / 256), 256, 0, stream>>>(
            xr, conv_w + (size_t)l * DI * 4, conv_b + (size_t)l * DI, xc);

        xproj_delta_kernel<<<(int)NT, DI, 0, stream>>>(
            xc, x_proj_w + (size_t)l * DI * XDBL, dt_proj_w + (size_t)l * DTR * DI,
            dt_proj_b + (size_t)l * DI, xdbl, delta);

        scan_kernel<<<(B_ * DI) / 8, 256, 0, stream>>>(
            delta, xc, xdbl, xr, A_log + (size_t)l * DI * NST,
            D_param + (size_t)l * DI, yg16);

        // out_proj: pack W[256,128] then GEMM [NT x 256]*[256 x 128] += h
        int totB = (DM/16) * (DI/32) * 32;             // 2048
        pack_w_kernel<<<(totB + 255) / 256, 256, 0, stream>>>(
            out_proj_w + (size_t)l * DI * DM, wfrag, DI, DM, totB);
        gemm_wmma_kernel<<<(int)((NT/16) * (DM/16) / 8), 256, 0, stream>>>(
            yg16, wfrag, h, DI, DM, 1);
    }

    final_kernel<<<(int)NT, DM, 0, stream>>>(h, out_norm_w, out_w, mean, stdv,
                                             (float*)d_out);
}
